// Encoder_14791867367468
// MI455X (gfx1250) — matrix operand order repaired and checked by reading, not compiled, not run on hardware
//
#include <hip/hip_runtime.h>
#include <hip/hip_bf16.h>
#include <math.h>

#define DIM 32
#define KCHUNKS 65              // 64 outer-product chunks + 1 bias chunk (K = 2080)
#define T2_ELEMS (KCHUNKS * 2 * 32 * 16)   // 66560 bf16 fragment elements

typedef __attribute__((ext_vector_type(16))) __bf16 v16bf;
typedef __attribute__((ext_vector_type(8)))  float  v8f;

__device__ __forceinline__ float sigf(float x) { return 1.0f / (1.0f + expf(-x)); }

// ---------------------------------------------------------------- utilities
__global__ void zero_f32(float* __restrict__ p, int n) {
    int i = blockIdx.x * 256 + threadIdx.x;
    if (i < n) p[i] = 0.0f;
}

__global__ void deg_kernel(const int* __restrict__ ei, float* __restrict__ deg, int E) {
    int e = blockIdx.x * 256 + threadIdx.x;
    if (e < E) atomicAdd(&deg[ei[E + e]], 1.0f);
}

// ------------------------------------------------- pack T2 into WMMA-B order
// Fragment element index: ((c*2 + t)*32 + lane)*16 + j
// lane l holds column n = t*16 + (l&15), K_local = ((l&16)?16:0) + j  (j=0..15)
__global__ void pack_kernel(const float* __restrict__ w2, const float* __restrict__ b2,
                            __bf16* __restrict__ T2) {
    int idx = blockIdx.x * 256 + threadIdx.x;
    if (idx >= T2_ELEMS) return;
    int j  = idx & 15;
    int l  = (idx >> 4) & 31;
    int ct = idx >> 9;          // c*2 + t
    int t  = ct & 1;
    int c  = ct >> 1;
    int k  = c * 32 + ((l & 16) ? 16 : 0) + j;
    int n  = t * 16 + (l & 15);
    float v;
    if (k < 2048) v = w2[(k >> 5) * 1024 + (k & 31) * 32 + n];   // w2[h, i*32+n]
    else          v = b2[(k - 2048) * 32 + n];                   // b2[i*32+n]
    T2[idx] = (__bf16)v;
}

// ---------------------------------------------------------------- lin0+relu
__global__ void lin0_kernel(const float* __restrict__ x, const float* __restrict__ w,
                            const float* __restrict__ b, float* __restrict__ out, int N) {
    __shared__ float sW[16 * 32];
    __shared__ float sB[32];
    int tid = threadIdx.x;
    for (int i = tid; i < 512; i += 256) sW[i] = w[i];
    if (tid < 32) sB[tid] = b[tid];
    __syncthreads();
    int idx = blockIdx.x * 256 + tid;
    if (idx >= N * 32) return;
    int n = idx >> 5, o = idx & 31;
    float s = sB[o];
    const float* xr = x + (long)n * 16;
#pragma unroll
    for (int i = 0; i < 16; ++i) s = fmaf(xr[i], sW[i * 32 + o], s);
    out[idx] = fmaxf(s, 0.0f);
}

// ----------------------------------------------- NNConv message pass (WMMA)
// Per wave: TWO tiles of 16 edges (M=32). msg = U[32x2080] @ T2[2080x32].
// A fragment for chunk c built in-registers as hid_e[c] * x_e[cols];
// B fragments double-buffered from LDS so ds_load latency hides behind WMMA.
__global__ void msg_kernel(const float* __restrict__ hin, const int* __restrict__ ei,
                           const float* __restrict__ ea, const float* __restrict__ w1,
                           const float* __restrict__ b1p, const __bf16* __restrict__ T2,
                           float* __restrict__ agg, int E, int tiles) {
    extern __shared__ char smem[];
    __bf16* sT2 = (__bf16*)smem;                       // 66560 bf16 = 133120 B
    float*  sW1 = (float*)(smem + T2_ELEMS * 2);       // 5*64
    float*  sB1 = sW1 + 320;                           // 64
    int tid = threadIdx.x;

    // cooperative copy of B fragments + edge-MLP layer-1 weights into LDS
    {
        float4* d4 = (float4*)sT2;
        const float4* s4 = (const float4*)T2;
        for (int i = tid; i < (T2_ELEMS * 2) / 16; i += 256) d4[i] = s4[i];
        for (int i = tid; i < 320; i += 256) sW1[i] = w1[i];
        if (tid < 64) sB1[tid] = b1p[tid];
    }
    __syncthreads();

    int wave  = tid >> 5, lane = tid & 31;
    int tbase = (blockIdx.x * 8 + wave) * 2;           // two consecutive tiles per wave
    if (tbase >= tiles) return;

    // A k order used here (contiguous k, same as packed B): lane<16 -> row=lane, K in {0..15}; lane>=16 -> K in {16..31}
    int row = lane & 15;
    int cb  = (lane & 16) ? 16 : 0;

    float xf[2][16];
    float eav[2][5];
#pragma unroll
    for (int tt = 0; tt < 2; ++tt) {
        int e   = (tbase + tt) * 16 + row;
        int eld = (e < E) ? e : (E - 1);
        int src = ei[eld];
        const float* xr = hin + (long)src * DIM;
#pragma unroll
        for (int j = 0; j < 16; ++j) {
            xf[tt][j] = xr[cb + j];
        }
#pragma unroll
        for (int j = 0; j < 5; ++j) eav[tt][j] = ea[(long)eld * 5 + j];
    }

    v8f acc[2][2] = {};
    const v16bf* bp = (const v16bf*)sT2;
    v16bf b0 = bp[lane];
    v16bf b1 = bp[32 + lane];

    for (int c = 0; c < KCHUNKS; ++c) {
        // build A fragments for both edge tiles
        v16bf a[2];
#pragma unroll
        for (int tt = 0; tt < 2; ++tt) {
            float hv;
            if (c < 64) {                               // hid_e[c] = relu(ea.w1[:,c]+b1[c])
                float s = sB1[c];
#pragma unroll
                for (int j = 0; j < 5; ++j) s = fmaf(eav[tt][j], sW1[j * 64 + c], s);
                hv = fmaxf(s, 0.0f);
            } else {
                hv = 1.0f;                              // bias chunk
            }
#pragma unroll
            for (int j = 0; j < 16; ++j) a[tt][j] = (__bf16)(xf[tt][j] * hv);
        }

        // prefetch next chunk's B fragments (double buffer)
        v16bf nb0 = b0, nb1 = b1;
        if (c + 1 < KCHUNKS) {
            nb0 = bp[(c + 1) * 64 + lane];
            nb1 = bp[(c + 1) * 64 + 32 + lane];
        }

        acc[0][0] = __builtin_amdgcn_wmma_f32_16x16x32_bf16(false, a[0], false, b0,
                                                            (short)0, acc[0][0], false, false);
        acc[0][1] = __builtin_amdgcn_wmma_f32_16x16x32_bf16(false, a[0], false, b1,
                                                            (short)0, acc[0][1], false, false);
        acc[1][0] = __builtin_amdgcn_wmma_f32_16x16x32_bf16(false, a[1], false, b0,
                                                            (short)0, acc[1][0], false, false);
        acc[1][1] = __builtin_amdgcn_wmma_f32_16x16x32_bf16(false, a[1], false, b1,
                                                            (short)0, acc[1][1], false, false);
        b0 = nb0;
        b1 = nb1;
    }

    // D-layout: lane holds col n=(lane&15)(+16 for t=1), rows mbase+v (v=VGPR idx)
    int ncol  = lane & 15;
    int mbase = (lane & 16) ? 8 : 0;
#pragma unroll
    for (int tt = 0; tt < 2; ++tt) {
        int tl = tbase + tt;
        if (tl >= tiles) continue;
#pragma unroll
        for (int v = 0; v < 8; ++v) {
            int eo = tl * 16 + mbase + v;
            if (eo < E) {
                int dst = ei[E + eo];
                atomicAdd(&agg[(long)dst * DIM + ncol],      acc[tt][0][v]);
                atomicAdd(&agg[(long)dst * DIM + 16 + ncol], acc[tt][1][v]);
            }
        }
    }
}

// --------------------------------------------- scatter-mean + bias + relu + GRU
__global__ void gru_kernel(const float* __restrict__ agg, const float* __restrict__ deg,
                           const float* __restrict__ hin, float* __restrict__ hout,
                           const float* __restrict__ wih, const float* __restrict__ whh,
                           const float* __restrict__ bih, const float* __restrict__ bhh,
                           const float* __restrict__ cb, int N) {
    __shared__ float sWih[96 * 32], sWhh[96 * 32], sBih[96], sBhh[96], sCb[32];
    int tid = threadIdx.x;
    for (int i = tid; i < 96 * 32; i += 256) { sWih[i] = wih[i]; sWhh[i] = whh[i]; }
    if (tid < 96) { sBih[tid] = bih[tid]; sBhh[tid] = bhh[tid]; }
    if (tid < 32) sCb[tid] = cb[tid];
    __syncthreads();
    int idx = blockIdx.x * 256 + tid;
    if (idx >= N * 32) return;
    int n = idx >> 5, o = idx & 31;
    const float* ar = agg + (long)n * 32;
    const float* hr = hin + (long)n * 32;
    float dinv = 1.0f / fmaxf(deg[n], 1.0f);
    float ir = sBih[o], iz = sBih[32 + o], inn = sBih[64 + o];
    float hrr = sBhh[o], hz = sBhh[32 + o], hn = sBhh[64 + o];
#pragma unroll
    for (int i = 0; i < 32; ++i) {
        float mi = fmaxf(fmaf(ar[i], dinv, sCb[i]), 0.0f);
        float hi = hr[i];
        ir  = fmaf(mi, sWih[o * 32 + i], ir);
        iz  = fmaf(mi, sWih[(32 + o) * 32 + i], iz);
        inn = fmaf(mi, sWih[(64 + o) * 32 + i], inn);
        hrr = fmaf(hi, sWhh[o * 32 + i], hrr);
        hz  = fmaf(hi, sWhh[(32 + o) * 32 + i], hz);
        hn  = fmaf(hi, sWhh[(64 + o) * 32 + i], hn);
    }
    float r  = sigf(ir + hrr);
    float z  = sigf(iz + hz);
    float nn = tanhf(inn + r * hn);
    hout[idx] = (1.0f - z) * nn + z * hr[o];
}

// ------------------------------------------------------------- Set2Set LSTM
__global__ void lstm_kernel(const float* __restrict__ qstar, float* __restrict__ hhb,
                            float* __restrict__ ccb, const float* __restrict__ wih,
                            const float* __restrict__ whh, const float* __restrict__ bih,
                            const float* __restrict__ bhh) {
    __shared__ float qs[64], sh[32];
    int b = blockIdx.x, o = threadIdx.x;       // 64 blocks x 32 threads
    qs[o] = qstar[b * 64 + o];
    qs[32 + o] = qstar[b * 64 + 32 + o];
    sh[o] = hhb[b * 32 + o];
    __syncthreads();
    float g[4];
#pragma unroll
    for (int r = 0; r < 4; ++r) {
        int j = r * 32 + o;
        float s = bih[j] + bhh[j];
        for (int k = 0; k < 64; ++k) s = fmaf(qs[k], wih[j * 64 + k], s);
        for (int k = 0; k < 32; ++k) s = fmaf(sh[k], whh[j * 32 + k], s);
        g[r] = s;
    }
    float i_ = sigf(g[0]), f_ = sigf(g[1]), g_ = tanhf(g[2]), o_ = sigf(g[3]);
    float c = f_ * ccb[b * 32 + o] + i_ * g_;
    ccb[b * 32 + o] = c;
    hhb[b * 32 + o] = o_ * tanhf(c);
}

// --------------------------------------- segment softmax attention + pooling
__device__ __forceinline__ unsigned enc_f32(float f) {
    unsigned u = __float_as_uint(f);
    return (u & 0x80000000u) ? ~u : (u | 0x80000000u);
}
__device__ __forceinline__ float dec_f32(unsigned v) {
    unsigned u = (v & 0x80000000u) ? (v & 0x7FFFFFFFu) : ~v;
    return __uint_as_float(u);
}

__global__ void attn_e_kernel(const float* __restrict__ feat, const float* __restrict__ q,
                              const int* __restrict__ batch, float* __restrict__ ebuf,
                              unsigned* __restrict__ emax, int N) {
    int n = blockIdx.x * 256 + threadIdx.x;
    if (n >= N) return;
    int g = batch[n];
    const float* fr = feat + (long)n * 32;
    const float* qr = q + g * 32;
    float s = 0.0f;
#pragma unroll
    for (int i = 0; i < 32; ++i) s = fmaf(fr[i], qr[i], s);
    ebuf[n] = s;
    atomicMax(&emax[g], enc_f32(s));
}

__global__ void attn_exp_kernel(float* __restrict__ ebuf, const unsigned* __restrict__ emax,
                                const int* __restrict__ batch, float* __restrict__ denom, int N) {
    int n = blockIdx.x * 256 + threadIdx.x;
    if (n >= N) return;
    int g = batch[n];
    float w = expf(ebuf[n] - dec_f32(emax[g]));
    ebuf[n] = w;
    atomicAdd(&denom[g], w);
}

__global__ void attn_pool_kernel(const float* __restrict__ ebuf, const float* __restrict__ denom,
                                 const int* __restrict__ batch, const float* __restrict__ feat,
                                 float* __restrict__ rpool, int N) {
    int idx = blockIdx.x * 256 + threadIdx.x;
    if (idx >= N * 32) return;
    int n = idx >> 5, i = idx & 31;
    int g = batch[n];
    float w = ebuf[n] / denom[g];
    atomicAdd(&rpool[g * 32 + i], w * feat[idx]);
}

__global__ void qstar_kernel(const float* __restrict__ hhb, const float* __restrict__ rpool,
                             float* __restrict__ qstar) {
    int idx = blockIdx.x * 256 + threadIdx.x;
    if (idx >= 4096) return;
    int b = idx >> 6, k = idx & 63;
    qstar[idx] = (k < 32) ? hhb[b * 32 + k] : rpool[b * 32 + (k - 32)];
}

// ------------------------------------------------------------------- driver
extern "C" void kernel_launch(void* const* d_in, const int* in_sizes, int n_in,
                              void* d_out, int out_size, void* d_ws, size_t ws_size,
                              hipStream_t stream) {
    const float* x       = (const float*)d_in[0];
    const int*   ei      = (const int*)d_in[1];
    const float* ea      = (const float*)d_in[2];
    const int*   batch   = (const int*)d_in[3];
    const float* lin0_w  = (const float*)d_in[4];
    const float* lin0_b  = (const float*)d_in[5];
    const float* mlp_w1  = (const float*)d_in[6];
    const float* mlp_b1  = (const float*)d_in[7];
    const float* mlp_w2  = (const float*)d_in[8];
    const float* mlp_b2  = (const float*)d_in[9];
    const float* cbias   = (const float*)d_in[10];
    const float* gwih    = (const float*)d_in[11];
    const float* gwhh    = (const float*)d_in[12];
    const float* gbih    = (const float*)d_in[13];
    const float* gbhh    = (const float*)d_in[14];
    const float* lwih    = (const float*)d_in[15];
    const float* lwhh    = (const float*)d_in[16];
    const float* lbih    = (const float*)d_in[17];
    const float* lbhh    = (const float*)d_in[18];

    int N = in_sizes[0] / 16;
    int E = in_sizes[2] / 5;
    int NV = N * 32;

    float* out   = (float*)d_out;
    float* qstar = out;            // [64*64]
    float* hB    = out + 4096;     // feat_last region [N*32]

    float*    ws    = (float*)d_ws;
    __bf16*   T2    = (__bf16*)ws;                  // 66560 bf16 (33280 f32 slots)
    float*    hA    = ws + 33280;
    float*    agg   = hA + NV;
    float*    deg   = agg + NV;
    float*    ebuf  = deg + N;
    float*    denom = ebuf + N;
    unsigned* emax  = (unsigned*)(denom + 64);
    float*    rpool = denom + 128;
    float*    hhb   = rpool + 2048;
    float*    ccb   = hhb + 2048;

    int tiles = (E + 15) / 16;
    int pairs = (tiles + 1) / 2;                      // 2 tiles per wave
    size_t msgShm = (size_t)T2_ELEMS * 2 + 384 * 4;   // B fragments + w1/b1

    // one-time (per call) precompute
    pack_kernel<<<(T2_ELEMS + 255) / 256, 256, 0, stream>>>(mlp_w2, mlp_b2, T2);
    lin0_kernel<<<(NV + 255) / 256, 256, 0, stream>>>(x, lin0_w, lin0_b, hA, N);
    zero_f32<<<(N + 255) / 256, 256, 0, stream>>>(deg, N);
    deg_kernel<<<(E + 255) / 256, 256, 0, stream>>>(ei, deg, E);

    // 3 message-passing + GRU iterations (ping-pong hA <-> hB)
    float* hin = hA;
    float* hout = hB;
    for (int it = 0; it < 3; ++it) {
        zero_f32<<<(NV + 255) / 256, 256, 0, stream>>>(agg, NV);
        msg_kernel<<<(pairs + 7) / 8, 256, msgShm, stream>>>(hin, ei, ea, mlp_w1, mlp_b1,
                                                             T2, agg, E, tiles);
        gru_kernel<<<(NV + 255) / 256, 256, 0, stream>>>(agg, deg, hin, hout,
                                                         gwih, gwhh, gbih, gbhh, cbias, N);
        float* t = hin; hin = hout; hout = t;
    }
    const float* feat = hin;   // == hB after 3 iterations (feat_last lives in d_out)

    // Set2Set (3 steps)
    zero_f32<<<16, 256, 0, stream>>>(qstar, 4096);
    zero_f32<<<8, 256, 0, stream>>>(hhb, 2048);
    zero_f32<<<8, 256, 0, stream>>>(ccb, 2048);
    for (int s = 0; s < 3; ++s) {
        lstm_kernel<<<64, 32, 0, stream>>>(qstar, hhb, ccb, lwih, lwhh, lbih, lbhh);
        zero_f32<<<1, 256, 0, stream>>>((float*)emax, 64);
        zero_f32<<<1, 256, 0, stream>>>(denom, 64);
        zero_f32<<<8, 256, 0, stream>>>(rpool, 2048);
        attn_e_kernel<<<(N + 255) / 256, 256, 0, stream>>>(feat, hhb, batch, ebuf, emax, N);
        attn_exp_kernel<<<(N + 255) / 256, 256, 0, stream>>>(ebuf, emax, batch, denom, N);
        attn_pool_kernel<<<(NV + 255) / 256, 256, 0, stream>>>(ebuf, denom, batch, feat, rpool, N);
        qstar_kernel<<<16, 256, 0, stream>>>(hhb, rpool, qstar);
    }
}
